// ScaledDotProductAttention_47820165873993
// MI455X (gfx1250) — matrix operand order repaired and checked by reading, not compiled, not run on hardware
//
#include <hip/hip_runtime.h>
#include <math.h>

// Flash attention for B=4,H=16,S=2048,D=128 (fp32 in/out), f16 WMMA compute,
// fp32 accumulation, online softmax. One block = 8 wave32 waves = 128 query
// rows of one (b,h). Scores are computed TRANSPOSED (S^T = K * Q^T) so the
// exp(P) matrix converts from the f32 C-layout into the f16 A-fragment for
// the P*V WMMA with per-lane packs only (layouts per ISA 7.12.2).
//
// v2: LDS double-buffering with one barrier per iteration; next block's
// global_load_b128s issued before compute (latency hidden behind WMMAs);
// all LDS fragments preloaded ahead of each WMMA chain so s_wait_dscnt
// can retire incrementally instead of 0x0 before every WMMA.

typedef __attribute__((ext_vector_type(16))) _Float16 v16h;
typedef __attribute__((ext_vector_type(8)))  _Float16 v8h;
typedef __attribute__((ext_vector_type(8)))  float    v8f;

static __device__ __forceinline__ v16h cat8(v8h lo, v8h hi) {
  return __builtin_shufflevector(lo, hi, 0,1,2,3,4,5,6,7,8,9,10,11,12,13,14,15);
}

static __device__ __forceinline__ v8h cvt8(float4 a, float4 b) {
  v8h r;
  r[0]=(_Float16)a.x; r[1]=(_Float16)a.y; r[2]=(_Float16)a.z; r[3]=(_Float16)a.w;
  r[4]=(_Float16)b.x; r[5]=(_Float16)b.y; r[6]=(_Float16)b.z; r[7]=(_Float16)b.w;
  return r;
}

__global__ __launch_bounds__(256)
void fa_fwd_kernel(const float* __restrict__ Q, const float* __restrict__ K,
                   const float* __restrict__ V, const int* __restrict__ MSK,
                   float* __restrict__ O)
{
  constexpr int S = 2048, D = 128;
  constexpr int KP = 136;  // sK pitch (halfs): 272B/row -> 16B aligned, bank-spread
  constexpr int VP = 40;   // sVT pitch (halfs): 80B/row -> 16B aligned, bank-spread

  __shared__ _Float16 sK [2][32  * KP];  // K block  [k=32][d=128], row-major f16
  __shared__ _Float16 sVT[2][128 * VP];  // V block transposed [d=128][k=32] f16

  const int tid  = threadIdx.x;
  const int wave = tid >> 5;
  const int lane = tid & 31;
  const int lh   = lane >> 4;     // wave half (0/1)
  const int l16  = lane & 15;
  const int bh   = blockIdx.y;    // b*16 + h
  const int b    = bh >> 4;       // H == 16
  const int q0   = blockIdx.x * 128 + wave * 16;

  const float* Qb = Q   + (size_t)bh * S * D;
  const float* Kb = K   + (size_t)bh * S * D;
  const float* Vb = V   + (size_t)bh * S * D;
  const int*   Mb = MSK + (size_t)b  * S;
  float*       Ob = O   + (size_t)bh * S * D;

  // ---- persistent Q^T B-fragments (32d x 16q per chunk): lane=q, elems=d ----
  v16h qf[4];
  {
    const float* qp = Qb + (size_t)(q0 + l16) * D + lh * 8;
#pragma unroll
    for (int c = 0; c < 4; ++c) {
      const float4* p = (const float4*)(qp + c * 32);
      qf[c] = cat8(cvt8(p[0], p[1]), cvt8(p[4], p[5]));
    }
  }

  v8f of[8];
#pragma unroll
  for (int t = 0; t < 8; ++t) of[t] = {};
  float m_run = -3.0e38f, l_run = 0.0f;
  const float scale = 0.08838834764831845f;   // 1/sqrt(128)

  const int kcp = tid >> 3;   // 0..31: row for cooperative K/V staging
  const int dg  = tid & 7;    // 0..7 : 16-wide d group

  // ---- software-pipelined staging: issue global loads, commit to LDS later --
  float4 kr0, kr1, kr2, kr3, vr0, vr1, vr2, vr3;
  auto stage_issue = [&](int kb) {
    const float4* kp = (const float4*)(Kb + (size_t)(kb + kcp) * D + dg * 16);
    kr0 = kp[0]; kr1 = kp[1]; kr2 = kp[2]; kr3 = kp[3];
    const float4* vp = (const float4*)(Vb + (size_t)(kb + kcp) * D + dg * 16);
    vr0 = vp[0]; vr1 = vp[1]; vr2 = vp[2]; vr3 = vp[3];
  };
  auto stage_commit = [&](int pb) {
    _Float16* dk = &sK[pb][kcp * KP + dg * 16];
    *(v8h*)(dk)     = cvt8(kr0, kr1);
    *(v8h*)(dk + 8) = cvt8(kr2, kr3);
    const float vv[16] = {vr0.x,vr0.y,vr0.z,vr0.w, vr1.x,vr1.y,vr1.z,vr1.w,
                          vr2.x,vr2.y,vr2.z,vr2.w, vr3.x,vr3.y,vr3.z,vr3.w};
#pragma unroll
    for (int j = 0; j < 16; ++j)
      sVT[pb][(dg * 16 + j) * VP + kcp] = (_Float16)vv[j];
  };

  int p = 0;
  stage_issue(0);
  stage_commit(0);
  __syncthreads();

  for (int kblk = 0; kblk < S; kblk += 32) {
    const bool more = (kblk + 32) < S;
    if (more) stage_issue(kblk + 32);   // in flight during this block's WMMAs

    // -------- preload all K A-fragments, then S^T = K * Q^T -----------------
    v16h ka[8];
#pragma unroll
    for (int c = 0; c < 4; ++c) {
      const int base = c * 32 + lh * 8;   // A-frag: elems 0-7 -> d=base+0..7
      const _Float16* k0p = &sK[p][ l16       * KP + base];
      const _Float16* k1p = &sK[p][(16 + l16) * KP + base];
      ka[2*c]   = cat8(*(const v8h*)(k0p), *(const v8h*)(k0p + 16));
      ka[2*c+1] = cat8(*(const v8h*)(k1p), *(const v8h*)(k1p + 16));
    }
    v8f c0 = {}, c1 = {};
#pragma unroll
    for (int c = 0; c < 4; ++c) {
      c0 = __builtin_amdgcn_wmma_f32_16x16x32_f16(false, ka[2*c],
             false, qf[c], (short)0, c0, false, false);
      c1 = __builtin_amdgcn_wmma_f32_16x16x32_f16(false, ka[2*c+1],
             false, qf[c], (short)0, c1, false, false);
    }

    // -------- mask + online softmax (row = q = lane%16 for stats) ----------
    int4 ma0 = *(const int4*)(Mb + kblk +      lh * 8);
    int4 ma1 = *(const int4*)(Mb + kblk +      lh * 8 + 4);
    int4 mb0 = *(const int4*)(Mb + kblk + 16 + lh * 8);
    int4 mb1 = *(const int4*)(Mb + kblk + 16 + lh * 8 + 4);
    const int mk0[8] = {ma0.x,ma0.y,ma0.z,ma0.w, ma1.x,ma1.y,ma1.z,ma1.w};
    const int mk1[8] = {mb0.x,mb0.y,mb0.z,mb0.w, mb1.x,mb1.y,mb1.z,mb1.w};

    float s0[8], s1[8];
    float mloc = -3.0e38f;
#pragma unroll
    for (int i = 0; i < 8; ++i) {
      s0[i] = mk0[i] ? c0[i] * scale : -1.0e9f;
      s1[i] = mk1[i] ? c1[i] * scale : -1.0e9f;
      mloc = fmaxf(mloc, fmaxf(s0[i], s1[i]));
    }
    mloc = fmaxf(mloc, __shfl_xor(mloc, 16, 32));   // combine k-halves per q
    const float mnew  = fmaxf(m_run, mloc);
    const float alpha = __expf(m_run - mnew);

    v16h pA;                                        // P A-frag (16q x 32k) f16
    float ssum = 0.0f;
#pragma unroll
    for (int i = 0; i < 8; ++i) {
      float p0 = __expf(s0[i] - mnew);
      float p1 = __expf(s1[i] - mnew);
      ssum += p0 + p1;
      pA[i]     = (_Float16)p0;
      pA[i + 8] = (_Float16)p1;
    }
    ssum += __shfl_xor(ssum, 16, 32);
    l_run = l_run * alpha + ssum;
    m_run = mnew;

    // alpha broadcast: O elem i holds row q = i + 8*lh; alpha for q sits in lane q
    float av[8];
#pragma unroll
    for (int i = 0; i < 8; ++i) av[i] = __shfl(alpha, i + 8 * lh, 32);
#pragma unroll
    for (int t = 0; t < 8; ++t)
#pragma unroll
      for (int i = 0; i < 8; ++i) of[t][i] *= av[i];

    // -------- preload all V B-fragments, then O += P * V --------------------
    v16h vb[8];
#pragma unroll
    for (int dt = 0; dt < 8; ++dt) {
      const _Float16* vp = &sVT[p][(dt * 16 + l16) * VP + lh * 8];
      vb[dt] = cat8(*(const v8h*)(vp), *(const v8h*)(vp + 16));
    }
#pragma unroll
    for (int dt = 0; dt < 8; ++dt)
      of[dt] = __builtin_amdgcn_wmma_f32_16x16x32_f16(false, pA,
                 false, vb[dt], (short)0, of[dt], false, false);

    // -------- commit next block into the other buffer; single barrier -------
    if (more) stage_commit(p ^ 1);
    __syncthreads();
    p ^= 1;
  }

  // -------- normalize by l and store --------------------------------------
  const float rl = 1.0f / l_run;
  float lv[8];
#pragma unroll
  for (int i = 0; i < 8; ++i) lv[i] = __shfl(rl, i + 8 * lh, 32);
#pragma unroll
  for (int dt = 0; dt < 8; ++dt)
#pragma unroll
    for (int i = 0; i < 8; ++i)
      Ob[(size_t)(q0 + i + 8 * lh) * D + dt * 16 + l16] = of[dt][i] * lv[i];
}

extern "C" void kernel_launch(void* const* d_in, const int* in_sizes, int n_in,
                              void* d_out, int out_size, void* d_ws, size_t ws_size,
                              hipStream_t stream) {
  (void)in_sizes; (void)n_in; (void)out_size; (void)d_ws; (void)ws_size;
  const float* Q   = (const float*)d_in[0];
  const float* K   = (const float*)d_in[1];
  const float* V   = (const float*)d_in[2];
  const int*   MSK = (const int*)d_in[3];
  float*       O   = (float*)d_out;
  // B=4, H=16, S=2048 -> grid (S/128, B*H)
  dim3 grid(2048 / 128, 4 * 16);
  fa_fwd_kernel<<<grid, 256, 0, stream>>>(Q, K, V, MSK, O);
}
